// GraphConvolutionLayer_63041529970794
// MI455X (gfx1250) — matrix operand order repaired and checked by reading, not compiled, not run on hardware
//
#include <hip/hip_runtime.h>

// WMMA operand vector types (probe-confirmed signatures for gfx1250)
typedef __attribute__((ext_vector_type(16))) __bf16 v16bf;
typedef __attribute__((ext_vector_type(8)))  float  v8f;
typedef __attribute__((ext_vector_type(4)))  float  v4f;   // native vector for NT stores

namespace {
constexpr int BT     = 24;      // B*T
constexpr int N_PTS  = 20000;
constexpr int KNB    = 16;      // neighbors
constexpr int CIN    = 32;
constexpr int COUT   = 64;
constexpr int TILE_M = 32;      // n-rows per block (two WMMA M tiles)
constexpr int BT_PER = 4;       // bt slices per block
constexpr int NTHREADS = 256;   // 8 wave32
constexpr float SIGMA2_INV = 1.0f / (0.6f * 0.6f);
}

// float -> bf16, round-to-nearest-even
static __device__ __forceinline__ unsigned short f2bf(float f) {
  union { float f; unsigned u; } v; v.f = f;
  unsigned u = v.u;
  return (unsigned short)((u + 0x7FFFu + ((u >> 16) & 1u)) >> 16);
}

// ---- pre-kernel: W (CIN,COUT) f32 -> W^T (COUT,CIN) bf16, once, into d_ws ----
__global__ __launch_bounds__(NTHREADS)
void prep_wt_kernel(const float* __restrict__ W, unsigned short* __restrict__ wt) {
  for (int i = threadIdx.x; i < COUT * CIN; i += NTHREADS) {
    const int n = i >> 5, k = i & 31;          // wt[n][k] = W[k][n]
    wt[i] = f2bf(W[k * COUT + n]);
  }
}

__global__ __launch_bounds__(NTHREADS)
void graphconv_wmma_kernel(const float* __restrict__ x,          // (BT, N, CIN)
                           const int*   __restrict__ nbr_idx,    // (N, KNB)
                           const float* __restrict__ dists,      // (N, KNB)
                           const unsigned short* __restrict__ wt,// (COUT, CIN) bf16
                           const float* __restrict__ bvec,       // (COUT)
                           float*       __restrict__ out)        // (BT, N, COUT)
{
  __shared__ int            sIdx[TILE_M][KNB];
  __shared__ float          sWgt[TILE_M][KNB];
  __shared__ __align__(64) unsigned short sA [TILE_M][CIN];  // bf16 A tile, row-major
  __shared__ __align__(64) unsigned short sWt[COUT][CIN];    // bf16 W^T
  __shared__ float          sOut[TILE_M][COUT];

  const int tid = threadIdx.x;
  const int n0  = blockIdx.x * TILE_M;
  const int bt0 = blockIdx.y * BT_PER;

  // ---- stage neighbor indices + Gaussian weights (exp once per n-tile) ----
  for (int i = tid; i < TILE_M * KNB; i += NTHREADS) {
    const int r = i >> 4, k = i & 15;
    const int g = (n0 + r) * KNB + k;
    sIdx[r][k] = nbr_idx[g];
    const float d = dists[g];
    sWgt[r][k] = __expf(-d * d * SIGMA2_INV);
  }
  // ---- stage pre-converted bf16 W^T (pure 4KB copy) ----
  {
    unsigned*       dst = reinterpret_cast<unsigned*>(&sWt[0][0]);
    const unsigned* src = reinterpret_cast<const unsigned*>(wt);
    for (int i = tid; i < (COUT * CIN) / 2; i += NTHREADS) dst[i] = src[i];
  }
  __syncthreads();

  // ---- hoisted wave-level WMMA constants (loop-invariant across bt) ----
  const int wv   = tid >> 5;        // wave id: rt = wv>>2 (row tile), ct = wv&3 (col tile)
  const int lane = tid & 31;
  const int hl   = lane >> 4;       // 0: lanes 0-15, 1: lanes 16-31
  const int lm   = lane & 15;
  const int rt   = wv >> 2;
  const int col  = (wv & 3) * 16 + lm;   // output column (B/C/D N index)
  const int arow = rt * 16 + lm;         // A-matrix row for this lane

  // B operand, this kernel's contiguous k order (not the ISA table order): lanes 0-15 hold K=0..15 of column `col`,
  // lanes 16-31 hold K=16..31 -> contiguous 32B from a W^T row.
  const v16bf bm = *reinterpret_cast<const v16bf*>(&sWt[col][hl * 16]);
  const float bias = bvec[col];

  // gather mapping: thread -> (row = tid/8 in 0..31, channels 4cg..4cg+3)
  const int grow = tid >> 3;
  const int cg   = tid & 7;

  for (int it = 0; it < BT_PER; ++it) {
    const int bt = bt0 + it;

    // ---- gather + weighted reduction in fp32 ----
    {
      const float4* __restrict__ xs =
          reinterpret_cast<const float4*>(x) + (size_t)bt * N_PTS * (CIN / 4);
      float4 acc = make_float4(0.f, 0.f, 0.f, 0.f);
#pragma unroll
      for (int k = 0; k < KNB; ++k) {
        const int   nb = sIdx[grow][k];
        const float wg = sWgt[grow][k];
        const float4 v = xs[(size_t)nb * (CIN / 4) + cg];
        acc.x = fmaf(wg, v.x, acc.x);
        acc.y = fmaf(wg, v.y, acc.y);
        acc.z = fmaf(wg, v.z, acc.z);
        acc.w = fmaf(wg, v.w, acc.w);
      }
      const unsigned lo = ((unsigned)f2bf(acc.y) << 16) | (unsigned)f2bf(acc.x);
      const unsigned hi = ((unsigned)f2bf(acc.w) << 16) | (unsigned)f2bf(acc.z);
      uint2 packed; packed.x = lo; packed.y = hi;
      *reinterpret_cast<uint2*>(&sA[grow][cg * 4]) = packed;
    }
    __syncthreads();

    // ---- WMMA: A(16x32 bf16) @ B(32x64 bf16) + bias, ReLU ----
    {
      // A operand, this kernel's contiguous k order (not the ISA table order):
      //   lanes 0-15 : K = 0..15,  lanes 16-31 : K = 16..31 (matches B)
      union { struct { uint4 lo, hi; } q; v16bf v; } ua;
      ua.q.lo = *reinterpret_cast<const uint4*>(&sA[arow][hl * 16]);
      ua.q.hi = *reinterpret_cast<const uint4*>(&sA[arow][hl * 16 + 8]);

      v8f c;
#pragma unroll
      for (int i = 0; i < 8; ++i) c[i] = bias;

      c = __builtin_amdgcn_wmma_f32_16x16x32_bf16(
          /*neg_a=*/false, ua.v, /*neg_b=*/false, bm,
          /*c_mod=*/(short)0, c, /*reuse_a=*/false, /*reuse_b=*/false);

      // ReLU + stage D: VGPR r holds M = r + 8*hl at N = col
#pragma unroll
      for (int r = 0; r < 8; ++r)
        sOut[rt * 16 + r + 8 * hl][col] = fmaxf(c[r], 0.0f);
    }
    __syncthreads();

    // ---- coalesced non-temporal store: 32x64 tile, 8 floats per thread ----
    // Output (123 MB) is write-once; NT stores keep x resident in the 192MB L2.
    {
      const int base = tid * 8;      // < 2048
      const int r    = base >> 6;
      const int cc   = base & 63;
      const v4f* s = reinterpret_cast<const v4f*>(&sOut[r][cc]);
      v4f* g = reinterpret_cast<v4f*>(
          out + ((size_t)bt * N_PTS + n0 + r) * COUT + cc);
      __builtin_nontemporal_store(s[0], g + 0);
      __builtin_nontemporal_store(s[1], g + 1);
    }
    // sA/sOut reuse hazards are covered by the two barriers above (next-iteration
    // gather writes happen only after this thread passes sync#2, and all sA reads
    // for this iteration complete before any thread passes sync#2).
  }
}

extern "C" void kernel_launch(void* const* d_in, const int* in_sizes, int n_in,
                              void* d_out, int out_size, void* d_ws, size_t ws_size,
                              hipStream_t stream) {
  (void)in_sizes; (void)n_in; (void)out_size; (void)ws_size;
  const float* x    = (const float*)d_in[0];  // (B,T,N,CIN) f32
  const int*   idx  = (const int*)  d_in[1];  // (N,K) int
  const float* dst  = (const float*)d_in[2];  // (N,K) f32
  const float* W    = (const float*)d_in[3];  // (CIN,COUT) f32
  const float* bias = (const float*)d_in[4];  // (COUT) f32
  float* out        = (float*)d_out;          // (B,T,N,COUT) f32

  unsigned short* wtBf = (unsigned short*)d_ws;  // 4KB bf16 W^T scratch

  prep_wt_kernel<<<1, NTHREADS, 0, stream>>>(W, wtBf);

  dim3 grid(N_PTS / TILE_M, BT / BT_PER);   // (625, 6)
  dim3 block(NTHREADS);                     // 8 wave32
  graphconv_wmma_kernel<<<grid, block, 0, stream>>>(x, idx, dst, wtBf, bias, out);
}
